// QuantLinear_43654047597182
// MI455X (gfx1250) — matrix operand order repaired and checked by reading, not compiled, not run on hardware
//
#include <hip/hip_runtime.h>
#include <hip/hip_bf16.h>
#include <stdint.h>

typedef __attribute__((ext_vector_type(16))) _Float16 v16h;
typedef __attribute__((ext_vector_type(8)))  _Float16 v8h;
typedef __attribute__((ext_vector_type(4)))  _Float16 v4h;
typedef __attribute__((ext_vector_type(2)))  _Float16 v2h;
typedef __attribute__((ext_vector_type(8)))  float    v8f;

#define K_DIM 11008
#define N_DIM 4096
#define GROUP 64

#define MBLK 128
#define NBLK 64
#define KBLK 32
#define LDSK 40   /* KBLK + 8 halves padding -> 80B row stride, 16B aligned */

// ---------------- async copy (global -> LDS), guarded ----------------
#if __has_builtin(__builtin_amdgcn_global_load_async_to_lds_b128)
#define HAVE_ASYNC 1
typedef int v4i_vec __attribute__((vector_size(16)));
typedef __attribute__((address_space(1))) v4i_vec* gvec_ptr;
typedef __attribute__((address_space(3))) v4i_vec* lvec_ptr;
__device__ __forceinline__ void async_copy16(const _Float16* g, _Float16* l) {
  __builtin_amdgcn_global_load_async_to_lds_b128(
      (gvec_ptr)(void*)(g), (lvec_ptr)(void*)(l), 0, 0);
}
#if __has_builtin(__builtin_amdgcn_s_wait_asynccnt)
__device__ __forceinline__ void async_wait0() { __builtin_amdgcn_s_wait_asynccnt(0); }
#else
__device__ __forceinline__ void async_wait0() { asm volatile("s_wait_asynccnt 0" ::: "memory"); }
#endif
#else
#define HAVE_ASYNC 0
__device__ __forceinline__ void async_copy16(const _Float16* g, _Float16* l) {
  *(v8h*)l = *(const v8h*)g;
}
__device__ __forceinline__ void async_wait0() {}
#endif

// ---------------- kernel 1: group-wise 4-bit fake-quant dequant -> f16 ----
// One wave32 per group of 64 contiguous weights. Lane holds 2 elements.
__global__ __launch_bounds__(256) void quant_dequant_w_kernel(
    const float* __restrict__ w, _Float16* __restrict__ wq, int ngroups) {
  const int wave = threadIdx.x >> 5;
  const int lane = threadIdx.x & 31;
  const int g = blockIdx.x * 8 + wave;
  if (g >= ngroups) return;

  const size_t base = (size_t)g * GROUP + (size_t)lane * 2;
  const float2 e = *(const float2*)(w + base);

  float mn = fminf(e.x, e.y);
  float mx = fmaxf(e.x, e.y);
#pragma unroll
  for (int off = 16; off >= 1; off >>= 1) {
    mn = fminf(mn, __shfl_xor(mn, off, 32));
    mx = fmaxf(mx, __shfl_xor(mx, off, 32));
  }
  const float scale = fmaxf(mx - mn, 1e-5f) * (1.0f / 15.0f);
  const float inv_s = 1.0f / scale;
  const float zp = rintf(-mn * inv_s);

  const float q0 = fminf(fmaxf(rintf(e.x * inv_s) + zp, 0.0f), 15.0f);
  const float q1 = fminf(fmaxf(rintf(e.y * inv_s) + zp, 0.0f), 15.0f);
  v2h o;
  o[0] = (_Float16)((q0 - zp) * scale);
  o[1] = (_Float16)((q1 - zp) * scale);
  *(v2h*)(wq + base) = o;
}

// ---------------- kernel 2: x fp32 -> f16 ----------------
__global__ __launch_bounds__(256) void cvt_x_kernel(
    const float* __restrict__ x, _Float16* __restrict__ y, int n4) {
  const int i = blockIdx.x * 256 + threadIdx.x;
  if (i >= n4) return;
  const float4 v = ((const float4*)x)[i];
  v4h o;
  o[0] = (_Float16)v.x; o[1] = (_Float16)v.y;
  o[2] = (_Float16)v.z; o[3] = (_Float16)v.w;
  ((v4h*)y)[i] = o;
}

// ---------------- kernel 3: WMMA GEMM: out = Xf16 * Wqf16^T + bias -------
// Block = 256 threads = 8 waves in 4(M) x 2(N); block tile 128(M) x 64(N),
// K step 32. Wave tile 32x32 = 2x2 WMMA 16x16 tiles. Grid 64 x 4 = 256 WGs.
__global__ __launch_bounds__(256) void wmma_gemm_kernel(
    const _Float16* __restrict__ X, const _Float16* __restrict__ W,
    const float* __restrict__ bias, float* __restrict__ out,
    int M, int N, int K) {
  __shared__ __align__(16) _Float16 lA[2][MBLK * LDSK];
  __shared__ __align__(16) _Float16 lB[2][NBLK * LDSK];

  const int tid  = threadIdx.x;
  const int wave = tid >> 5;
  const int lane = tid & 31;
  const int wm = wave >> 1;  // 0..3 -> 32 rows each
  const int wn = wave & 1;   // 0..1 -> 32 cols each
  const int m0 = blockIdx.y * MBLK;
  const int n0 = blockIdx.x * NBLK;

  v8f acc[2][2];
#pragma unroll
  for (int i = 0; i < 2; ++i)
#pragma unroll
    for (int j = 0; j < 2; ++j) acc[i][j] = (v8f)(0.0f);

  // A tile: 128x32 halves = 512 16B-chunks (2/thread); B tile: 64x32 = 256 (1/thread).
  auto load_tiles = [&](int buf, int kblk) {
#pragma unroll
    for (int i = 0; i < 2; ++i) {
      const int c  = tid + 256 * i;
      const int r  = c >> 2;          // 4 chunks per 32-half row
      const int kc = (c & 3) * 8;     // halves
      async_copy16(X + (size_t)(m0 + r) * K + kblk + kc, &lA[buf][r * LDSK + kc]);
    }
    {
      const int r  = tid >> 2;
      const int kc = (tid & 3) * 8;
      async_copy16(W + (size_t)(n0 + r) * K + kblk + kc, &lB[buf][r * LDSK + kc]);
    }
  };

  auto compute = [&](int buf) {
    const _Float16* As = lA[buf];
    const _Float16* Bs = lB[buf];
    const int lr = lane & 15;   // M (for A) / N (for B)
    const int kh = lane >> 4;   // half-wave selects K halves
    v16h a[2], b[2];
#pragma unroll
    for (int i = 0; i < 2; ++i) {
      const int row = wm * 32 + i * 16 + lr;
      const v8h lo = *(const v8h*)&As[row * LDSK + kh * 8];        // K 0..7 / 8..15
      const v8h hi = *(const v8h*)&As[row * LDSK + 16 + kh * 8];   // K 16..23 / 24..31
      a[i] = __builtin_shufflevector(lo, hi, 0, 1, 2, 3, 4, 5, 6, 7,
                                     8, 9, 10, 11, 12, 13, 14, 15);
    }
#pragma unroll
    for (int j = 0; j < 2; ++j) {
      const int col = wn * 32 + j * 16 + lr;
      const v8h lo = *(const v8h*)&Bs[col * LDSK + kh * 8];        // K 0..7 / 8..15
      const v8h hi = *(const v8h*)&Bs[col * LDSK + 16 + kh * 8];   // K 16..23 / 24..31
      b[j] = __builtin_shufflevector(lo, hi, 0, 1, 2, 3, 4, 5, 6, 7,
                                     8, 9, 10, 11, 12, 13, 14, 15);
    }
#pragma unroll
    for (int i = 0; i < 2; ++i)
#pragma unroll
      for (int j = 0; j < 2; ++j)
        acc[i][j] = __builtin_amdgcn_wmma_f32_16x16x32_f16(
            false, a[i], false, b[j], (short)0, acc[i][j], false, false);
  };

  const int NT = K / KBLK;
  load_tiles(0, 0);
  int cur = 0;
  for (int t = 0; t < NT; ++t) {
    async_wait0();
    __syncthreads();                       // staged tile visible to all waves
    if (t + 1 < NT) {                      // prefetch next tile, overlap compute
      load_tiles(cur ^ 1, (t + 1) * KBLK);
    }
    compute(cur);
    __syncthreads();                       // all reads done before next overwrite
    cur ^= 1;
  }

  // Epilogue: C/D layout — VGPR v: lanes 0-15 -> M=v, lanes 16-31 -> M=v+8.
  const int lr = lane & 15;
  const int hh = lane >> 4;
#pragma unroll
  for (int j = 0; j < 2; ++j) {
    const int col = n0 + wn * 32 + j * 16 + lr;
    const float bv = bias[col];
#pragma unroll
    for (int i = 0; i < 2; ++i) {
#pragma unroll
      for (int v = 0; v < 8; ++v) {
        const int row = m0 + wm * 32 + i * 16 + v + hh * 8;
        out[(size_t)row * N + col] = acc[i][j][v] + bv;
      }
    }
  }
}

extern "C" void kernel_launch(void* const* d_in, const int* in_sizes, int n_in,
                              void* d_out, int out_size, void* d_ws, size_t ws_size,
                              hipStream_t stream) {
  const float* x    = (const float*)d_in[0];   // [1, 512, 11008] fp32
  const float* w    = (const float*)d_in[1];   // [4096, 11008] fp32
  const float* bias = (const float*)d_in[2];   // [4096] fp32
  float* out = (float*)d_out;                  // [1, 512, 4096] fp32

  const int K = K_DIM;
  const int N = N_DIM;
  const int M = in_sizes[0] / K;               // 512

  // Workspace layout: [Wq f16 : N*K][X f16 : M*K]
  _Float16* wq  = (_Float16*)d_ws;
  _Float16* x16 = (_Float16*)((char*)d_ws + (size_t)N * K * sizeof(_Float16));

  // 1) dequantize weight -> f16 (one wave32 per group of 64)
  const int ngroups = (N * K) / GROUP;         // 704512
  quant_dequant_w_kernel<<<(ngroups + 7) / 8, 256, 0, stream>>>(w, wq, ngroups);

  // 2) x fp32 -> f16
  const int n4 = (M * K) / 4;
  cvt_x_kernel<<<(n4 + 255) / 256, 256, 0, stream>>>(x, x16, n4);

  // 3) WMMA GEMM + bias
  dim3 grid(N / NBLK, M / MBLK);               // 64 x 4 = 256 workgroups
  wmma_gemm_kernel<<<grid, 256, 0, stream>>>(x16, wq, bias, out, M, N, K);
}
